// GIN_72937134621131
// MI455X (gfx1250) — matrix operand order repaired and checked by reading, not compiled, not run on hardware
//
#include <hip/hip_runtime.h>
#include <hip/hip_bf16.h>

// ---------------------------------------------------------------------------
// GIN on MI455X (gfx1250): HBM-bound edge scatter + WMMA bf16 MLPs.
// K is fixed at 128 for every GEMM, so each wave holds its entire 16-row A
// strip (32 VGPRs) and sweeps all N tiles -> A read exactly once from HBM.
// GEMM epilogue is template-specialized (ReLU->bf16 vs linear->f32) so the
// hot loop around v_wmma_f32_16x16x32_bf16 is branch-free.
// ---------------------------------------------------------------------------

#define N_NODES 50000
#define N_EDGES 600000
#define HID     128
#define OUTF    64
#define N_LAYERS 4
#define N_GRAPHS 128
#define BN_EPS  1e-5f

typedef __attribute__((ext_vector_type(16))) __bf16 v16bf;
typedef __attribute__((ext_vector_type(8)))  float  v8f;

union ABReg { unsigned int u[8]; v16bf v; };

__device__ __forceinline__ unsigned short f2bf(float f) {
    unsigned int u = __float_as_uint(f);
    unsigned int r = u + 0x7FFFu + ((u >> 16) & 1u);   // round-to-nearest-even
    return (unsigned short)(r >> 16);
}

// ----------------------------- utility kernels ------------------------------

__global__ void zero_f32(float* __restrict__ p, int n) {
    int i = blockIdx.x * blockDim.x + threadIdx.x;
    if (i < n) p[i] = 0.0f;
}

__global__ void f32_to_bf16(const float* __restrict__ in, unsigned short* __restrict__ out, int n) {
    int i = blockIdx.x * blockDim.x + threadIdx.x;
    if (i < n) out[i] = f2bf(in[i]);
}

// hB[i] = bf16(h[i] + agg[i])   (GIN eps=0 combine, fused with A-operand cast)
__global__ void add_to_bf16(const float* __restrict__ h, const float* __restrict__ agg,
                            unsigned short* __restrict__ out, int n) {
    int i = blockIdx.x * blockDim.x + threadIdx.x;
    if (i < n) out[i] = f2bf(h[i] + agg[i]);
}

// ------------------------------ edge scatter --------------------------------
// one lane handles (edge, 4 features): float4 gather + 4 global f32 atomics.
__global__ void edge_scatter(const float* __restrict__ h, const int* __restrict__ src,
                             const int* __restrict__ dst, float* __restrict__ agg, int nEdges) {
    long long idx = (long long)blockIdx.x * blockDim.x + threadIdx.x;
    int e = (int)(idx >> 5);
    if (e >= nEdges) return;
    int q = ((int)idx & 31) * 4;
    int s = src[e], d = dst[e];
    const float4 v = *(const float4*)(h + (size_t)s * HID + q);
    float* p = agg + (size_t)d * HID + q;
    atomicAdd(p + 0, v.x); atomicAdd(p + 1, v.y);
    atomicAdd(p + 2, v.z); atomicAdd(p + 3, v.w);
}

// ------------------------- weight pack (B operand) --------------------------
// W is [K x N] f32 row-major.  Packed layout: Bp[((nt*ntK + kt)*32 + lane)*8 + v]
// lane 0-15 -> K = kt*32 + {0..7,16..23}, lane 16-31 -> K = kt*32 + {8..15,24..31}; N = nt*16 + (lane&15)
// dword v packs bf16(K=kb+2v') | bf16(K=kb+2v'+1)<<16, kb = kt*32+(lane>>4)*8+(v>>2)*16, v'=v&3 (matches A table order)
__global__ void pack_b(const float* __restrict__ W, unsigned int* __restrict__ Bp, int K, int N) {
    int idx = blockIdx.x * blockDim.x + threadIdx.x;
    int total = (K >> 5) * (N >> 4) * 32 * 8;
    if (idx >= total) return;
    int v    = idx & 7;
    int lane = (idx >> 3) & 31;
    int t    = idx >> 8;
    int ntK  = K >> 5;
    int kt = t % ntK, nt = t / ntK;
    int k = kt * 32 + (lane >> 4) * 8 + (v >> 2) * 16 + 2 * (v & 3);
    int n = nt * 16 + (lane & 15);
    unsigned int lo = f2bf(W[(size_t)k * N + n]);
    unsigned int hi = f2bf(W[(size_t)(k + 1) * N + n]);
    Bp[idx] = lo | (hi << 16);
}

// ------------------------------- WMMA GEMM ----------------------------------
// Y[M,N] = act(A[M,128](bf16 row-major) @ Wpacked + bias),  K fixed at 128.
// One wave owns a 16-row strip: loads the full A strip into 32 VGPRs once,
// then sweeps all N tiles (v_wmma_f32_16x16x32_bf16, 4 per tile).  B tiles are
// tiny (<=32KB) and stay L2/WGP$ resident across the sweep.
// RELU: fuse max(0,.) into epilogue.  BF16OUT: store bf16 else f32.
// M, N multiples of 16.  Only wave-uniform control flow: EXEC all-ones at WMMA.
template <int RELU, int BF16OUT>
__global__ void gemm_bf16_wmma(const unsigned short* __restrict__ A,
                               const unsigned int*  __restrict__ Bp,
                               const float*         __restrict__ bias,
                               void*                __restrict__ outP,
                               int N, int numMTiles) {
    int lane = threadIdx.x & 31;
    int mt = blockIdx.x * (blockDim.x >> 5) + (threadIdx.x >> 5);
    if (mt >= numMTiles) return;               // wave-uniform
    int ntN = N >> 4;
    int hi = lane >> 4;
    const unsigned short* Arow = A + (size_t)(mt * 16 + (lane & 15)) * 128;

    // 16-bit A 16x32 layout (ISA 7.12.2): lane group hi, v<4 -> K=kb+2v,
    // v>=4 -> K=kb+16+2(v-4);  kb = kt*32 + hi*8.  All 4 K-tiles held at once.
    ABReg a[4];
    #pragma unroll
    for (int kt = 0; kt < 4; ++kt) {
        int kb = kt * 32 + hi * 8;
        #pragma unroll
        for (int v = 0; v < 4; ++v)
            a[kt].u[v] = *(const unsigned int*)(Arow + kb + 2 * v);
        #pragma unroll
        for (int v = 4; v < 8; ++v)
            a[kt].u[v] = *(const unsigned int*)(Arow + kb + 16 + 2 * (v - 4));
    }

    for (int nt = 0; nt < ntN; ++nt) {
        v8f c = {};
        #pragma unroll
        for (int kt = 0; kt < 4; ++kt) {
            ABReg b;
            const unsigned int* bsrc = Bp + ((size_t)(nt * 4 + kt) * 32 + lane) * 8;
            #pragma unroll
            for (int v = 0; v < 8; ++v) b.u[v] = bsrc[v];
            c = __builtin_amdgcn_wmma_f32_16x16x32_bf16(false, a[kt].v, false, b.v,
                                                        (short)0, c, false, false);
        }
        int n = nt * 16 + (lane & 15);
        float bv = bias[n];                     // one coalesced load per lane per tile
        size_t rowBase = (size_t)(mt * 16 + hi * 8) * N + n;
        #pragma unroll
        for (int r = 0; r < 8; ++r) {           // C layout: VGPR r -> M = r + hi*8
            float val = c[r] + bv;
            if (RELU) val = fmaxf(val, 0.0f);
            if (BF16OUT) ((unsigned short*)outP)[rowBase + (size_t)r * N] = f2bf(val);
            else         ((float*)outP)[rowBase + (size_t)r * N] = val;
        }
    }
}

// ------------------------------ batch norm ----------------------------------

__global__ void bn_stats(const float* __restrict__ h, float* __restrict__ sum,
                         float* __restrict__ sq, int rows, int rowsPerBlock) {
    __shared__ float ls[HID], lq[HID];
    int tid = threadIdx.x;                     // 256 threads
    if (tid < HID) { ls[tid] = 0.0f; lq[tid] = 0.0f; }
    __syncthreads();
    int f = tid & (HID - 1), half = tid >> 7;
    int start = blockIdx.x * rowsPerBlock;
    int end = start + rowsPerBlock; if (end > rows) end = rows;
    float s = 0.0f, q = 0.0f;
    for (int r = start + half; r < end; r += 2) {
        float v = h[(size_t)r * HID + f];
        s += v; q += v * v;
    }
    atomicAdd(&ls[f], s); atomicAdd(&lq[f], q);
    __syncthreads();
    if (tid < HID) { atomicAdd(&sum[tid], ls[tid]); atomicAdd(&sq[tid], lq[tid]); }
}

__global__ void bn_apply(float* __restrict__ h, const float* __restrict__ sum,
                         const float* __restrict__ sq, const float* __restrict__ gamma,
                         const float* __restrict__ beta, int rows) {
    int idx = blockIdx.x * blockDim.x + threadIdx.x;
    if (idx >= rows * HID) return;
    int f = idx & (HID - 1);
    float inv  = 1.0f / (float)rows;
    float mean = sum[f] * inv;
    float var  = sq[f] * inv - mean * mean;     // population var (ddof=0)
    float sc   = gamma[f] * rsqrtf(var + BN_EPS);
    float sh   = beta[f] - mean * sc;
    h[idx] = fmaxf(h[idx] * sc + sh, 0.0f);
}

// --------------------------------- pooling ----------------------------------

__global__ void pool_kernel(const float* __restrict__ h, const int* __restrict__ batch,
                            float* __restrict__ g, int rows) {
    long long idx = (long long)blockIdx.x * blockDim.x + threadIdx.x;
    int node = (int)(idx >> 5);
    if (node >= rows) return;
    int q = ((int)idx & 31) * 4;
    int gi = batch[node];
    const float4 v = *(const float4*)(h + (size_t)node * HID + q);
    float* p = g + (size_t)gi * HID + q;
    atomicAdd(p + 0, v.x); atomicAdd(p + 1, v.y);
    atomicAdd(p + 2, v.z); atomicAdd(p + 3, v.w);
}

// --------------------------------- launch -----------------------------------

extern "C" void kernel_launch(void* const* d_in, const int* in_sizes, int n_in,
                              void* d_out, int out_size, void* d_ws, size_t ws_size,
                              hipStream_t stream) {
    const float* x     = (const float*)d_in[0];
    const int*   edge  = (const int*)  d_in[1];
    const int*   batch = (const int*)  d_in[2];
    const float* W1    = (const float*)d_in[3];
    const float* b1    = (const float*)d_in[4];
    const float* W2    = (const float*)d_in[5];
    const float* b2    = (const float*)d_in[6];
    const float* gamma = (const float*)d_in[7];
    const float* beta  = (const float*)d_in[8];
    const float* Wf1   = (const float*)d_in[9];
    const float* bf1   = (const float*)d_in[10];
    const float* Wf2   = (const float*)d_in[11];
    const float* bf2   = (const float*)d_in[12];
    float* out = (float*)d_out;

    char* ws = (char*)d_ws;
    // 256B-aligned workspace carve-out (~77.3 MB total)
    float*          h     = (float*)(ws + 0);                          // 25,600,000 B
    float*          agg   = (float*)(ws + 25600000);                   // 25,600,000 B
    unsigned short* hB    = (unsigned short*)(ws + 51200000);          // 12,800,000 B
    unsigned short* tmpB  = (unsigned short*)(ws + 64000000);          // 12,800,000 B
    unsigned int*   W1p   = (unsigned int*)(ws + 76800000);            // 131,072 B (4 layers)
    unsigned int*   W2p   = (unsigned int*)(ws + 76931072);            // 131,072 B
    unsigned int*   Wf1p  = (unsigned int*)(ws + 77062144);            // 32,768 B
    unsigned int*   Wf2p  = (unsigned int*)(ws + 77094912);            // 16,384 B
    float*          stSum = (float*)(ws + 77111296);                   // 512 B
    float*          stSq  = (float*)(ws + 77111808);                   // 512 B
    float*          g     = (float*)(ws + 77112320);                   // 65,536 B
    unsigned short* gB    = (unsigned short*)(ws + 77177856);          // 32,768 B
    unsigned short* fc1B  = (unsigned short*)(ws + 77210624);          // 32,768 B

    const int* src = edge;
    const int* dst = edge + N_EDGES;

    const int NH = N_NODES * HID;                       // 6,400,000
    const int TPB = 256;

    // h <- x
    hipMemcpyAsync(h, x, (size_t)NH * sizeof(float), hipMemcpyDeviceToDevice, stream);

    // pack all weights into WMMA B-operand layout (bf16)
    for (int l = 0; l < N_LAYERS; ++l) {
        pack_b<<<(8192 + TPB - 1) / TPB, TPB, 0, stream>>>(W1 + (size_t)l * HID * HID, W1p + (size_t)l * 8192, HID, HID);
        pack_b<<<(8192 + TPB - 1) / TPB, TPB, 0, stream>>>(W2 + (size_t)l * HID * HID, W2p + (size_t)l * 8192, HID, HID);
    }
    pack_b<<<(8192 + TPB - 1) / TPB, TPB, 0, stream>>>(Wf1, Wf1p, HID, HID);
    pack_b<<<(4096 + TPB - 1) / TPB, TPB, 0, stream>>>(Wf2, Wf2p, HID, OUTF);

    const int mTiles  = N_NODES / 16;                   // 3125 (exact)
    const int mBlocks = (mTiles + 3) / 4;               // 4 waves / block (128 threads)

    for (int l = 0; l < N_LAYERS; ++l) {
        // agg = segment_sum(h[src], dst)
        zero_f32<<<(NH + TPB - 1) / TPB, TPB, 0, stream>>>(agg, NH);
        {
            long long nThreads = (long long)N_EDGES * 32;
            int blocks = (int)((nThreads + TPB - 1) / TPB);
            edge_scatter<<<blocks, TPB, 0, stream>>>(h, src, dst, agg, N_EDGES);
        }
        // hB = bf16(h + agg)
        add_to_bf16<<<(NH + TPB - 1) / TPB, TPB, 0, stream>>>(h, agg, hB, NH);
        // tmpB = bf16(relu(hB @ W1[l] + b1[l]))
        gemm_bf16_wmma<1, 1><<<mBlocks, 128, 0, stream>>>(hB, W1p + (size_t)l * 8192,
                                                          b1 + l * HID, tmpB, HID, mTiles);
        // h = tmpB @ W2[l] + b2[l]   (f32)
        gemm_bf16_wmma<0, 0><<<mBlocks, 128, 0, stream>>>(tmpB, W2p + (size_t)l * 8192,
                                                          b2 + l * HID, h, HID, mTiles);
        // BatchNorm (training-mode batch stats) + ReLU, in place
        zero_f32<<<1, 256, 0, stream>>>(stSum, HID);
        zero_f32<<<1, 256, 0, stream>>>(stSq, HID);
        {
            int rowsPerBlock = 256;
            int blocks = (N_NODES + rowsPerBlock - 1) / rowsPerBlock;
            bn_stats<<<blocks, 256, 0, stream>>>(h, stSum, stSq, N_NODES, rowsPerBlock);
        }
        bn_apply<<<(NH + TPB - 1) / TPB, TPB, 0, stream>>>(h, stSum, stSq,
                                                           gamma + l * HID, beta + l * HID, N_NODES);
    }

    // global_add_pool
    zero_f32<<<(N_GRAPHS * HID + TPB - 1) / TPB, TPB, 0, stream>>>(g, N_GRAPHS * HID);
    {
        long long nThreads = (long long)N_NODES * 32;
        int blocks = (int)((nThreads + TPB - 1) / TPB);
        pool_kernel<<<blocks, TPB, 0, stream>>>(h, batch, g, N_NODES);
    }
    f32_to_bf16<<<(N_GRAPHS * HID + TPB - 1) / TPB, TPB, 0, stream>>>(g, gB, N_GRAPHS * HID);

    // head: fc1B = bf16(relu(g @ Wf1 + bf1));  out = fc1B @ Wf2 + bf2
    {
        int mT = N_GRAPHS / 16;                         // 8
        gemm_bf16_wmma<1, 1><<<(mT + 3) / 4, 128, 0, stream>>>(gB, Wf1p, bf1, fc1B, HID, mT);
        gemm_bf16_wmma<0, 0><<<(mT + 3) / 4, 128, 0, stream>>>(fc1B, Wf2p, bf2, out, OUTF, mT);
    }
}
